// NgramMultiheadAttention_66048007078053
// MI455X (gfx1250) — matrix operand order repaired and checked by reading, not compiled, not run on hardware
//
#include <hip/hip_runtime.h>

// ---------------------------------------------------------------------------
// NgramMultiheadAttention for MI455X (gfx1250), wave32 + WMMA f16.
//
// Pipeline (all on `stream`, intermediates in d_ws, f16 wherever possible):
//   0) cvt_f16: one-shot f32->f16 of query / in_proj_w / rel_w / out_w
//      (everything is L2-resident; f16 halves operand re-read traffic and
//      removes all v_cvt from the GEMM inner loops).
//   1) gemm_nt<0>: qkv = q16 @ w_in^T + b -> f16 (q cols pre-scaled by
//      HD^-0.5), side-writes V^T [bh][dim][t] for the P@V WMMA B operand.
//   2) gemm_nt<1>: relvals = q16 @ rel_w^T + b -> f32 (6144 x 512).
//   3) attn_fused (main): flash attention per (bh, 16 q rows), causal +
//      bucketed rel bias; K/V tiles double-buffered into LDS with
//      global_load_async_to_lds_b128 / s_wait_asynccnt when available.
//   4) attn_fused (ngram x2): causal over main keys + diagonal predict tile.
//   5) gemm_nt<2>: out = attn @ out_w^T + out_b -> f32 d_out.
// ---------------------------------------------------------------------------

typedef __attribute__((ext_vector_type(16))) _Float16 v16h;
typedef __attribute__((ext_vector_type(8)))  float    v8f;
typedef __attribute__((ext_vector_type(4)))  int      v4i;

constexpr int T = 512, Bb = 4, E = 1024, H = 16, HD = 64, NGRAM = 2, NB = 32;
constexpr int TGT = (1 + NGRAM) * T;   // 1536
constexpr int E3  = 3 * E;             // 3072
constexpr int NBH = NB * H;            // 512
constexpr int MROWS = TGT * Bb;        // 6144 flattened (t, b) rows
constexpr float NEGV = -1.0e9f;
constexpr float SCAL = 0.125f;         // HD^-0.5

#define DEVINL __device__ __forceinline__
#define AS1 __attribute__((address_space(1)))
#define AS3 __attribute__((address_space(3)))

// CDNA5 async global->LDS path (ASYNCcnt), guarded so either toolchain builds.
#if defined(__has_builtin)
#  if __has_builtin(__builtin_amdgcn_global_load_async_to_lds_b128) && \
      __has_builtin(__builtin_amdgcn_s_wait_asynccnt)
#    define USE_ASYNC 1
#  endif
#endif
#ifndef USE_ASYNC
#  define USE_ASYNC 0
#endif

union V16 { v16h h; uint4 q[2]; };

DEVINL v8f vzero() { v8f z = {0.f,0.f,0.f,0.f,0.f,0.f,0.f,0.f}; return z; }

DEVINL v8f WMMA(v16h a, v16h b, v8f c) {
  return __builtin_amdgcn_wmma_f32_16x16x32_f16(
      false, a, false, b, (short)0, c, false, false);
}

// b128 async copy operands: pointer-to-int4 in AS1 (global) / AS3 (LDS).
DEVINL AS1 v4i* as1(const void* p) {
  return (AS1 v4i*)(unsigned long long)p;
}
DEVINL AS3 v4i* as3(const void* p) {
  // flat shared pointers carry the wave-relative LDS offset in the low 32 bits
  return (AS3 v4i*)(unsigned int)(unsigned long long)p;
}

// A fragment (16x32, per-lane row = lane&15): elems 0..15 = K[hi*16..+15].
// p = row pointer at k0 (16B aligned).
DEVINL v16h ldA_h(const _Float16* p, int hi) {
  V16 u;
  u.q[0] = *(const uint4*)(p + hi * 16);
  u.q[1] = *(const uint4*)(p + hi * 16 + 8);
  return u.h;
}
// B fragment (32x16, per-lane col = lane&15): 16 contiguous K values.
// Caller passes (column base + hi*16).
DEVINL v16h ldB_h(const _Float16* p) {
  V16 u;
  u.q[0] = *(const uint4*)(p);
  u.q[1] = *(const uint4*)(p + 8);
  return u.h;
}

// ---------------------------------------------------------------------------
// One-shot f32 -> f16 conversion (sizes here are multiples of 1024).
// ---------------------------------------------------------------------------
__global__ __launch_bounds__(256) void cvt_f16(const float* __restrict__ s,
                                               _Float16* __restrict__ d, int n) {
  const int i = (blockIdx.x * 256 + threadIdx.x) * 4;
  if (i + 3 < n) {
    const float4 v = *(const float4*)(s + i);
    union { _Float16 h[4]; uint2 u; } o;
    o.h[0] = (_Float16)v.x; o.h[1] = (_Float16)v.y;
    o.h[2] = (_Float16)v.z; o.h[3] = (_Float16)v.w;
    *(uint2*)(d + i) = o.u;
  } else {
    for (int j = 0; i + j < n && j < 4; ++j) d[i + j] = (_Float16)s[i + j];
  }
}

// ---------------------------------------------------------------------------
// NT GEMM (all-f16 operands):  C[M x N] = A[M x K] @ W[N x K]^T + bias.
// One wave per 64x64 C tile (4x4 WMMA register tile) -> good L2 reuse.
// OMODE 0: f16 out (qkv; cols < E scaled; cols >= 2E side-written to V^T).
// OMODE 1/2: f32 out.
// ---------------------------------------------------------------------------
template <int OMODE>
__global__ __launch_bounds__(32) void gemm_nt(
    const _Float16* __restrict__ A, const _Float16* __restrict__ W,
    const float* __restrict__ bias, void* __restrict__ Cp,
    _Float16* __restrict__ vT, int M, int N, int K)
{
  const int lane = threadIdx.x & 31;
  const int hi = lane >> 4, l15 = lane & 15;
  const int n0 = blockIdx.x * 64, m0 = blockIdx.y * 64;

  const _Float16* wb = W + (size_t)(n0 + l15) * K + hi * 16;

  v8f c[4][4];
#pragma unroll
  for (int mi = 0; mi < 4; ++mi)
#pragma unroll
    for (int ni = 0; ni < 4; ++ni) c[mi][ni] = vzero();

  for (int k0 = 0; k0 < K; k0 += 32) {
    if (k0 + 32 < K) {  // global_prefetch_b8 of the next A line
      __builtin_prefetch(A + (size_t)(m0 + l15) * K + k0 + 32, 0, 0);
    }
    v16h a[4], b[4];
#pragma unroll
    for (int mi = 0; mi < 4; ++mi)
      a[mi] = ldA_h(A + (size_t)(m0 + mi * 16 + l15) * K + k0, hi);
#pragma unroll
    for (int ni = 0; ni < 4; ++ni)
      b[ni] = ldB_h(wb + (size_t)ni * 16 * K + k0);
#pragma unroll
    for (int mi = 0; mi < 4; ++mi)
#pragma unroll
      for (int ni = 0; ni < 4; ++ni)
        c[mi][ni] = WMMA(a[mi], b[ni], c[mi][ni]);
  }

#pragma unroll
  for (int ni = 0; ni < 4; ++ni) {
    const int col = n0 + ni * 16 + l15;
    const float bv = bias[col];
#pragma unroll
    for (int mi = 0; mi < 4; ++mi) {
#pragma unroll
      for (int r = 0; r < 8; ++r) {
        const int row = m0 + mi * 16 + r + 8 * hi;   // C layout: VGPR r -> M=r(+8)
        float val = c[mi][ni][r] + bv;
        if constexpr (OMODE == 0) {
          _Float16* q16 = (_Float16*)Cp;
          const float sv = (col < E) ? val * SCAL : val;
          q16[(size_t)row * E3 + col] = (_Float16)sv;
          if (col >= 2 * E) {                         // side-write V^T
            const int jj = col - 2 * E;
            const int hh = jj >> 6, dd = jj & 63;
            const int t = row >> 2, bbx = row & 3;    // row = t*B + b, B=4
            vT[((size_t)((bbx << 4) + hh) * 64 + dd) * TGT + t] = (_Float16)val;
          }
        } else {
          ((float*)Cp)[(size_t)row * N + col] = val;
        }
      }
    }
  }
}

// ---------------------------------------------------------------------------
// Fused flash attention.  One wave per (16 query rows, bh[, ngram]).
// 32-key chunks: 4 WMMAs for S = Q K^T, bucketed rel-bias gather + mask +
// running-max softmax in the C layout, P transposed through LDS into the A
// layout, 4 WMMAs for P @ V (via V^T).  K/V tiles are double-buffered into
// LDS with async global->LDS loads when the toolchain exposes them.
// Single-wave workgroups => __syncthreads() is an S_NOP per the CDNA5 ISA.
// ---------------------------------------------------------------------------
__global__ __launch_bounds__(32) void attn_fused(
    const _Float16* __restrict__ qkv, const float* __restrict__ relv,
    const _Float16* __restrict__ vT, const int* __restrict__ buckets,
    _Float16* __restrict__ attnw, int is_pred)
{
  __shared__ _Float16 Pt[16 * 32];     // P tile (16 rows x 32 keys)
#if USE_ASYNC
  __shared__ _Float16 Kt[2][32 * 64];  // keys x dims, double buffered
  __shared__ _Float16 Vt[2][64 * 32];  // dims x keys (V^T tile), double buffered
#endif

  const int lane = threadIdx.x & 31;
  const int hi = lane >> 4, l15 = lane & 15;
  const int t0 = blockIdx.x * 16;
  const int bh = blockIdx.y;
  const int b = bh >> 4, h = bh & 15;
  const int n = blockIdx.z;
  const int qrow0 = is_pred ? (T + n * T) : 0;
  const int SW = is_pred ? (2 * T) : T;

  const _Float16* qp =
      qkv + (size_t)((qrow0 + t0 + l15) * Bb + b) * E3 + h * HD;
  const v16h aQ0 = ldA_h(qp, hi);
  const v16h aQ1 = ldA_h(qp + 32, hi);

  v8f acc0 = vzero(), acc1 = vzero(), acc2 = vzero(), acc3 = vzero();
  float rmax[8], rsum[8];
#pragma unroll
  for (int r = 0; r < 8; ++r) { rmax[r] = -3.0e38f; rsum[r] = 0.f; }

  const _Float16* vtb = vT + (size_t)bh * 64 * TGT;
  const int nchunk = (t0 + 16 + 31) / 32;                // causal key chunks
  const int total = nchunk + (is_pred ? 1 : 0);          // + diagonal tile

  auto growOf = [&](int c) -> int {                      // first key row of chunk c
    if (c < nchunk) return c * 32;
    const int ssx = (t0 <= T - 32) ? t0 : (T - 32);
    return T + n * T + ssx;
  };

#if USE_ASYNC
  auto stage = [&](int buf, int grow) {                  // 16 async b128 ops
    const _Float16* gk =
        qkv + (size_t)((grow + lane) * Bb + b) * E3 + E + h * HD;
    _Float16* lk = &Kt[buf][lane * 64];
#pragma unroll
    for (int i = 0; i < 8; ++i)
      __builtin_amdgcn_global_load_async_to_lds_b128(
          as1(gk + i * 8), as3(lk + i * 8), 0, 0);
    const _Float16* gv0 = vtb + (size_t)lane * TGT + grow;
    const _Float16* gv1 = vtb + (size_t)(lane + 32) * TGT + grow;
    _Float16* lv0 = &Vt[buf][lane * 32];
    _Float16* lv1 = &Vt[buf][(lane + 32) * 32];
#pragma unroll
    for (int i = 0; i < 4; ++i) {
      __builtin_amdgcn_global_load_async_to_lds_b128(
          as1(gv0 + i * 8), as3(lv0 + i * 8), 0, 0);
      __builtin_amdgcn_global_load_async_to_lds_b128(
          as1(gv1 + i * 8), as3(lv1 + i * 8), 0, 0);
    }
  };
  stage(0, growOf(0));
#endif

  for (int cidx = 0; cidx < total; ++cidx) {
    const bool diag = (cidx == nchunk);
    const int grow0 = growOf(cidx);
    const int ss = diag ? (grow0 - (T + n * T)) : 0;
    const int cb = cidx & 1;

#if USE_ASYNC
    if (cidx + 1 < total) {
      stage(cb ^ 1, growOf(cidx + 1));            // prefetch next chunk
      __builtin_amdgcn_s_wait_asynccnt(16);       // drain current chunk only
    } else {
      __builtin_amdgcn_s_wait_asynccnt(0);
    }
#endif

    // S = Q K^T for the two 16-key tiles of this 32-key chunk
    v8f s0t = vzero(), s1t = vzero();
#if USE_ASYNC
    {
      const _Float16* kb = &Kt[cb][0];
      s0t = WMMA(aQ0, ldB_h(kb + l15 * 64 + hi * 16), s0t);
      s0t = WMMA(aQ1, ldB_h(kb + l15 * 64 + 32 + hi * 16), s0t);
      s1t = WMMA(aQ0, ldB_h(kb + (16 + l15) * 64 + hi * 16), s1t);
      s1t = WMMA(aQ1, ldB_h(kb + (16 + l15) * 64 + 32 + hi * 16), s1t);
    }
#else
    {
      const _Float16* k0p =
          qkv + (size_t)((grow0 + l15) * Bb + b) * E3 + E + h * HD;
      s0t = WMMA(aQ0, ldB_h(k0p + hi * 16), s0t);
      s0t = WMMA(aQ1, ldB_h(k0p + 32 + hi * 16), s0t);
      const _Float16* k1p =
          qkv + (size_t)((grow0 + 16 + l15) * Bb + b) * E3 + E + h * HD;
      s1t = WMMA(aQ0, ldB_h(k1p + hi * 16), s1t);
      s1t = WMMA(aQ1, ldB_h(k1p + 32 + hi * 16), s1t);
    }
#endif

    __syncthreads();   // S_NOP (single-wave WG); fences the LDS P tile
#pragma unroll
    for (int r = 0; r < 8; ++r) {
      const int t = t0 + r + 8 * hi;                     // C layout row
      const size_t rr = (size_t)((qrow0 + t) * Bb + b);  // relvals row
      const int bl = (b * T + t) * SW;                   // bucket row base
      int sm0, sm1; float mk0, mk1;
      if (!diag) {
        sm0 = grow0 + l15;  sm1 = grow0 + 16 + l15;
        mk0 = (sm0 > t) ? NEGV : 0.f;
        mk1 = (sm1 > t) ? NEGV : 0.f;
      } else {
        const int k0i = ss + l15, k1i = ss + 16 + l15;
        sm0 = T + k0i;  sm1 = T + k1i;
        mk0 = (k0i == t) ? 0.f : NEGV;                   // NEG*(1-eye)
        mk1 = (k1i == t) ? 0.f : NEGV;
      }
      const float rel0 = relv[rr * NBH + (size_t)buckets[bl + sm0] * H + h];
      const float rel1 = relv[rr * NBH + (size_t)buckets[bl + sm1] * H + h];
      float v0 = s0t[r] + rel0 + mk0;
      float v1 = s1t[r] + rel1 + mk1;

      float m = fmaxf(fmaxf(v0, v1), rmax[r]);
#pragma unroll
      for (int w = 1; w < 16; w <<= 1) m = fmaxf(m, __shfl_xor(m, w, 32));
      const float f  = __expf(rmax[r] - m);
      const float p0 = __expf(v0 - m), p1 = __expf(v1 - m);
      float ps = p0 + p1;
#pragma unroll
      for (int w = 1; w < 16; w <<= 1) ps += __shfl_xor(ps, w, 32);
      rsum[r] = rsum[r] * f + ps;
      rmax[r] = m;
      acc0[r] *= f; acc1[r] *= f; acc2[r] *= f; acc3[r] *= f;

      Pt[(r + 8 * hi) * 32 + l15]      = (_Float16)p0;
      Pt[(r + 8 * hi) * 32 + 16 + l15] = (_Float16)p1;
    }
    __syncthreads();   // S_NOP; in-order DS pipeline makes P visible

    // P (A layout from LDS) @ V (B layout), 4 output dim tiles
    const v16h aP = ldA_h(&Pt[l15 * 32], hi);
#if USE_ASYNC
    const _Float16* vb = &Vt[cb][0];
    acc0 = WMMA(aP, ldB_h(vb + l15 * 32 + hi * 16), acc0);
    acc1 = WMMA(aP, ldB_h(vb + (16 + l15) * 32 + hi * 16), acc1);
    acc2 = WMMA(aP, ldB_h(vb + (32 + l15) * 32 + hi * 16), acc2);
    acc3 = WMMA(aP, ldB_h(vb + (48 + l15) * 32 + hi * 16), acc3);
#else
    const _Float16* vp = vtb + (size_t)l15 * TGT + grow0 + hi * 16;
    acc0 = WMMA(aP, ldB_h(vp),            acc0);
    acc1 = WMMA(aP, ldB_h(vp + 16 * TGT), acc1);
    acc2 = WMMA(aP, ldB_h(vp + 32 * TGT), acc2);
    acc3 = WMMA(aP, ldB_h(vp + 48 * TGT), acc3);
#endif
  }

  // Normalize and store attn rows (f16) for the out-projection GEMM.
#pragma unroll
  for (int r = 0; r < 8; ++r) {
    const int t = t0 + r + 8 * hi;
    const float inv = 1.0f / rsum[r];
    _Float16* op =
        attnw + (size_t)((qrow0 + t) * Bb + b) * E + h * HD + l15;
    op[0]  = (_Float16)(acc0[r] * inv);
    op[16] = (_Float16)(acc1[r] * inv);
    op[32] = (_Float16)(acc2[r] * inv);
    op[48] = (_Float16)(acc3[r] * inv);
  }
}

// ---------------------------------------------------------------------------
extern "C" void kernel_launch(void* const* d_in, const int* in_sizes, int n_in,
                              void* d_out, int out_size, void* d_ws, size_t ws_size,
                              hipStream_t stream) {
  (void)in_sizes; (void)n_in; (void)out_size; (void)ws_size;

  const float* query = (const float*)d_in[0];
  const float* ipw   = (const float*)d_in[1];
  const float* ipb   = (const float*)d_in[2];
  const float* relw  = (const float*)d_in[3];
  const float* relb  = (const float*)d_in[4];
  const float* outw  = (const float*)d_in[5];
  const float* outb  = (const float*)d_in[6];
  const int*   mainb = (const int*)d_in[7];
  const int*   predb = (const int*)d_in[8];
  float* out = (float*)d_out;

  // Workspace carve (~85 MB):
  _Float16* qkv  = (_Float16*)d_ws;                         // 6144 x 3072 f16
  float*    relv = (float*)(qkv + (size_t)MROWS * E3);      // 6144 x  512 f32
  _Float16* vT   = (_Float16*)(relv + (size_t)MROWS * NBH); // 64x64x1536  f16
  _Float16* qh   = vT + (size_t)64 * 64 * TGT;              // 6144 x 1024 f16
  _Float16* wih  = qh + (size_t)MROWS * E;                  // 3072 x 1024 f16
  _Float16* wrh  = wih + (size_t)E3 * E;                    //  512 x 1024 f16
  _Float16* woh  = wrh + (size_t)NBH * E;                   // 1024 x 1024 f16
  _Float16* attw = qh;   // alias: q16 is dead once both projections ran

  dim3 blk32(32), blk256(256);

  cvt_f16<<<(MROWS * E) / 1024, blk256, 0, stream>>>(query, qh, MROWS * E);
  cvt_f16<<<(E3 * E)    / 1024, blk256, 0, stream>>>(ipw,  wih, E3 * E);
  cvt_f16<<<(NBH * E)   / 1024, blk256, 0, stream>>>(relw, wrh, NBH * E);
  cvt_f16<<<(E * E)     / 1024, blk256, 0, stream>>>(outw, woh, E * E);

  gemm_nt<0><<<dim3(E3 / 64, MROWS / 64), blk32, 0, stream>>>(
      qh, wih, ipb, qkv, vT, MROWS, E3, E);
  gemm_nt<1><<<dim3(NBH / 64, MROWS / 64), blk32, 0, stream>>>(
      qh, wrh, relb, relv, nullptr, MROWS, NBH, E);

  attn_fused<<<dim3(T / 16, Bb * H, 1), blk32, 0, stream>>>(
      qkv, relv, vT, mainb, attw, 0);
  attn_fused<<<dim3(T / 16, Bb * H, NGRAM), blk32, 0, stream>>>(
      qkv, relv, vT, predb, attw, 1);

  gemm_nt<2><<<dim3(E / 64, MROWS / 64), blk32, 0, stream>>>(
      attw, woh, outb, out, nullptr, MROWS, E, E);
}
